// RMSU_Block_4226247819911
// MI455X (gfx1250) — matrix operand order repaired and checked by reading, not compiled, not run on hardware
//
#include <hip/hip_runtime.h>

#define LNUM   2
#define SDIM   64
#define DDIM   128
#define BATCH  128
#define TLEN   2000
#define MROWS  32            // batch rows per block (2 WMMA M-tiles)
#define NTHR   256           // 8 waves: (mtile 0..1) x (ntile 0..3)
#define ALPHA  0.1f
#define GAMMA  0.05f
#define WINDOW 100.0f
#define EPS    1e-5f

typedef __attribute__((ext_vector_type(16))) _Float16 v16h;
typedef __attribute__((ext_vector_type(8)))  _Float16 v8h;
typedef __attribute__((ext_vector_type(8)))  float    v8f;

// ---- LDS: all weights resident in f16 for the whole scan (CDNA5: 320KB/WGP) ----
struct Smem {
    // f32 tiles first (all sizes multiples of 16B -> alignment preserved)
    float pmu [MROWS * DDIM];        // running mu state (per-thread owned)
    float xn32[MROWS * DDIM];        // normalized x (residual source)
    float xo  [MROWS * DDIM];        // layer input / output x tile
    float lnp [LNUM * 4 * DDIM];     // ln_g, ln_b, bn_scale, bn_shift
    // f16 weights, row-major [N][K] == direct B-matrix (W^T) feed
    _Float16 wg[LNUM * SDIM * DDIM];
    _Float16 ug[LNUM * SDIM * SDIM];
    _Float16 wq[LNUM * SDIM * DDIM];
    _Float16 wv[LNUM * SDIM * DDIM];
    _Float16 bw[LNUM * SDIM * DDIM];
    _Float16 wo[LNUM * DDIM * SDIM];
    // f16 activation tiles (A-matrix feeds)
    _Float16 xa [MROWS * DDIM];
    _Float16 dxa[MROWS * DDIM];
    _Float16 st [LNUM * 2 * MROWS * SDIM];   // double-buffered recurrent state
};

// A-matrix 16x32 f16 fragment (ISA 7.12.2): lane m=lane&15, half=lane>>4,
// halves at [kb+8h, +8) and [kb+16+8h, +8)  -> 2x ds_load_b128
__device__ __forceinline__ v16h ld_a(const _Float16* t, int kdim, int kstep, int lane) {
    const int m = lane & 15, h = lane >> 4;
    const _Float16* p = t + m * kdim + kstep * 32 + h * 8;
    v16h a;
    ((v8h*)&a)[0] = *(const v8h*)(p);
    ((v8h*)&a)[1] = *(const v8h*)(p + 16);
    return a;
}

// B-matrix 32x16 f16 fragment: lane n = ntile*16 + (lane&15),
// halves at [kb+8h, +8) and [kb+16+8h, +8) -> 2x ds_load_b128
__device__ __forceinline__ v16h ld_b(const _Float16* w, int kdim, int ntile, int kstep, int lane) {
    const int n = ntile * 16 + (lane & 15), h = lane >> 4;
    const _Float16* p = w + n * kdim + kstep * 32 + h * 8;
    v16h b;
    ((v8h*)&b)[0] = *(const v8h*)(p);
    ((v8h*)&b)[1] = *(const v8h*)(p + 16);
    return b;
}

#define WMMA(A, Bm, C) \
    __builtin_amdgcn_wmma_f32_16x16x32_f16(false, (A), false, (Bm), (short)0, (C), false, false)

__device__ __forceinline__ float sigf(float x) { return 1.f / (1.f + __expf(-x)); }

__global__ void __launch_bounds__(NTHR, 1)
rmsu_scan_kernel(const float* __restrict__ x_seq,
                 const float* __restrict__ Wg_w, const float* __restrict__ Wg_b,
                 const float* __restrict__ Ug_w, const float* __restrict__ Ug_b,
                 const float* __restrict__ b_g,
                 const float* __restrict__ Wq_w, const float* __restrict__ Wq_b,
                 const float* __restrict__ Wv_w, const float* __restrict__ Wv_b,
                 const float* __restrict__ Avec,
                 const float* __restrict__ B_w,  const float* __restrict__ B_b,
                 const float* __restrict__ Wo_w, const float* __restrict__ Wo_b,
                 const float* __restrict__ ln_g, const float* __restrict__ ln_b,
                 const float* __restrict__ bn_g, const float* __restrict__ bn_b,
                 const float* __restrict__ bn_rm, const float* __restrict__ bn_rv,
                 float* __restrict__ out)
{
    extern __shared__ char smem_raw[];
    Smem& sm = *reinterpret_cast<Smem*>(smem_raw);

    const int tid   = threadIdx.x;
    const int lane  = tid & 31;
    const int wvid  = tid >> 5;        // 8 waves
    const int mt    = wvid >> 2;       // M-tile (batch rows mt*16..+16)
    const int w4    = wvid & 3;        // N-tile over S=64
    const int hf    = lane >> 4;
    const int l16   = lane & 15;
    const int brow0 = blockIdx.x * MROWS;

    // ---- one-time setup: weights f32 -> f16 into LDS ----
    for (int i = tid; i < LNUM * SDIM * DDIM; i += NTHR) sm.wg[i] = (_Float16)Wg_w[i];
    for (int i = tid; i < LNUM * SDIM * SDIM; i += NTHR) sm.ug[i] = (_Float16)Ug_w[i];
    for (int i = tid; i < LNUM * SDIM * DDIM; i += NTHR) sm.wq[i] = (_Float16)Wq_w[i];
    for (int i = tid; i < LNUM * SDIM * DDIM; i += NTHR) sm.wv[i] = (_Float16)Wv_w[i];
    for (int i = tid; i < LNUM * SDIM * DDIM; i += NTHR) sm.bw[i] = (_Float16)B_w[i];
    for (int i = tid; i < LNUM * DDIM * SDIM; i += NTHR) sm.wo[i] = (_Float16)Wo_w[i];
    for (int i = tid; i < LNUM * DDIM; i += NTHR) {
        const int l = i / DDIM, c = i % DDIM;
        const float sc = bn_g[i] * rsqrtf(bn_rv[i] + EPS);
        sm.lnp[(l * 4 + 0) * DDIM + c] = ln_g[i];
        sm.lnp[(l * 4 + 1) * DDIM + c] = ln_b[i];
        sm.lnp[(l * 4 + 2) * DDIM + c] = sc;
        sm.lnp[(l * 4 + 3) * DDIM + c] = bn_b[i] - bn_rm[i] * sc;
    }
    for (int i = tid; i < MROWS * DDIM; i += NTHR)           sm.pmu[i] = 0.f;
    for (int i = tid; i < LNUM * 2 * MROWS * SDIM; i += NTHR) sm.st[i] = (_Float16)0.f;
    __syncthreads();

    // ---- per-lane C-fragment constants (column n = w4*16 + l16) ----
    const int ncol = w4 * 16 + l16;
    float gb[LNUM], qb[LNUM], vb[LNUM], bbias[LNUM], tA[LNUM], wob[LNUM][2];
    #pragma unroll
    for (int l = 0; l < LNUM; ++l) {
        gb[l]    = Wg_b[l * SDIM + ncol] + Ug_b[l * SDIM + ncol] + b_g[l * SDIM + ncol];
        qb[l]    = Wq_b[l * SDIM + ncol];
        vb[l]    = Wv_b[l * SDIM + ncol];
        bbias[l] = B_b [l * SDIM + ncol];
        tA[l]    = tanhf(Avec[l * SDIM + ncol]);
        wob[l][0] = Wo_b[l * DDIM + (w4 * 2 + 0) * 16 + l16];
        wob[l][1] = Wo_b[l * DDIM + (w4 * 2 + 1) * 16 + l16];
    }
    const v8f zero8 = {0.f, 0.f, 0.f, 0.f, 0.f, 0.f, 0.f, 0.f};
    v8f ps[LNUM], pa[LNUM];             // recurrent state / attn, resident in VGPRs
    ps[0] = zero8; ps[1] = zero8; pa[0] = zero8; pa[1] = zero8;

    const int row = tid >> 3, sub = tid & 7;     // 8 lanes per batch row
    const size_t xbase = ((size_t)(brow0 + row) * TLEN) * DDIM + (size_t)(sub * 16);

    for (int t = 0; t < TLEN; ++t) {
        const int cb = t & 1, pb = cb ^ 1;       // state double-buffer indices

        // ---- stream x_t tile into LDS; prefetch next step (global_prefetch_b8) ----
        const float* xp = x_seq + xbase + (size_t)t * DDIM;
        float4 r0 = ((const float4*)xp)[0];
        float4 r1 = ((const float4*)xp)[1];
        float4 r2 = ((const float4*)xp)[2];
        float4 r3 = ((const float4*)xp)[3];
        {
            float* xod = &sm.xo[row * DDIM + sub * 16];
            ((float4*)xod)[0] = r0; ((float4*)xod)[1] = r1;
            ((float4*)xod)[2] = r2; ((float4*)xod)[3] = r3;
        }
        if (t + 1 < TLEN) __builtin_prefetch(xp + DDIM, 0, 0);
        __syncthreads();

        #pragma unroll
        for (int l = 0; l < LNUM; ++l) {
            // ================= LayerNorm + BatchNorm + mu update =================
            {
                float xv[16];
                const float* xr = &sm.xo[row * DDIM + sub * 16];
                #pragma unroll
                for (int i = 0; i < 4; ++i) {
                    float4 v = ((const float4*)xr)[i];
                    xv[i*4+0] = v.x; xv[i*4+1] = v.y; xv[i*4+2] = v.z; xv[i*4+3] = v.w;
                }
                float s = 0.f, ss = 0.f;
                #pragma unroll
                for (int i = 0; i < 16; ++i) { s += xv[i]; ss += xv[i] * xv[i]; }
                s += __shfl_xor(s, 1, 32); ss += __shfl_xor(ss, 1, 32);
                s += __shfl_xor(s, 2, 32); ss += __shfl_xor(ss, 2, 32);
                s += __shfl_xor(s, 4, 32); ss += __shfl_xor(ss, 4, 32);
                const float mean = s * (1.f / DDIM);
                const float var  = ss * (1.f / DDIM) - mean * mean;
                const float rstd = rsqrtf(var + EPS);
                #pragma unroll
                for (int i = 0; i < 16; ++i) {
                    const int c = sub * 16 + i;
                    float xn = (xv[i] - mean) * rstd * sm.lnp[(l*4+0)*DDIM + c]
                             + sm.lnp[(l*4+1)*DDIM + c];
                    xn = xn * sm.lnp[(l*4+2)*DDIM + c] + sm.lnp[(l*4+3)*DDIM + c];
                    const int idx = row * DDIM + c;
                    sm.xa[idx]   = (_Float16)xn;
                    sm.xn32[idx] = xn;
                    const float pm = sm.pmu[idx];
                    const float dx = xn - pm;          // delta_mu * WINDOW
                    sm.dxa[idx]  = (_Float16)dx;
                    sm.pmu[idx]  = pm + dx * (1.f / WINDOW);
                }
            }
            __syncthreads();

            // ================= WMMA phase: software-pipelined fragments =================
            const _Float16* xat = sm.xa  + mt * 16 * DDIM;
            const _Float16* dxt = sm.dxa + mt * 16 * DDIM;
            const _Float16* stp = sm.st + ((l * 2 + pb) * MROWS + mt * 16) * SDIM; // prev state
            const _Float16* wgl = sm.wg + l * SDIM * DDIM;
            const _Float16* wql = sm.wq + l * SDIM * DDIM;
            const _Float16* wvl = sm.wv + l * SDIM * DDIM;
            const _Float16* bwl = sm.bw + l * SDIM * DDIM;
            const _Float16* ugl = sm.ug + l * SDIM * SDIM;

            // hoisted Ug operands (consumed last -> long latency tolerance)
            v16h au0 = ld_a(stp, SDIM, 0, lane);
            v16h au1 = ld_a(stp, SDIM, 1, lane);
            v16h bu0 = ld_b(ugl, SDIM, w4, 0, lane);
            v16h bu1 = ld_b(ugl, SDIM, w4, 1, lane);

            // 2-stage pipeline over k: stage k+1 loads in flight during stage k wmmas
            v16h axs[2], dxs[2], stg[2][4];
            axs[0]    = ld_a(xat, DDIM, 0, lane);
            dxs[0]    = ld_a(dxt, DDIM, 0, lane);
            stg[0][0] = ld_b(wgl, DDIM, w4, 0, lane);
            stg[0][1] = ld_b(wql, DDIM, w4, 0, lane);
            stg[0][2] = ld_b(wvl, DDIM, w4, 0, lane);
            stg[0][3] = ld_b(bwl, DDIM, w4, 0, lane);

            v8f accg = zero8, accq = zero8, accv = zero8, accb = zero8, accd = zero8;
            #pragma unroll
            for (int k = 0; k < 4; ++k) {
                const int cu = k & 1, nx = cu ^ 1;
                if (k < 3) {
                    axs[nx]    = ld_a(xat, DDIM, k + 1, lane);
                    dxs[nx]    = ld_a(dxt, DDIM, k + 1, lane);
                    stg[nx][0] = ld_b(wgl, DDIM, w4, k + 1, lane);
                    stg[nx][1] = ld_b(wql, DDIM, w4, k + 1, lane);
                    stg[nx][2] = ld_b(wvl, DDIM, w4, k + 1, lane);
                    stg[nx][3] = ld_b(bwl, DDIM, w4, k + 1, lane);
                }
                accg = WMMA(axs[cu], stg[cu][0], accg);
                accq = WMMA(axs[cu], stg[cu][1], accq);
                accv = WMMA(axs[cu], stg[cu][2], accv);
                accb = WMMA(axs[cu], stg[cu][3], accb);
                accd = WMMA(dxs[cu], stg[cu][3], accd);
            }
            accg = WMMA(au0, bu0, accg);
            accg = WMMA(au1, bu1, accg);

            // ================= pointwise recurrence on C-fragments =================
            #pragma unroll
            for (int r = 0; r < 8; ++r) {
                const float g  = sigf(accg[r] + gb[l]);
                const float q  = accq[r] + qb[l];
                const float vv = accv[r] + vb[l];
                const float at = q * vv + __expf(-ALPHA * g) * pa[l][r];
                const float sn = tA[l] * ps[l][r]
                               + (accb[r] + bbias[l]) * (1.f - g)
                               + (GAMMA / WINDOW) * accd[r]
                               + at;
                pa[l][r] = at;
                ps[l][r] = sn;
            }
            // write curr state into the OTHER buffer: no barrier needed before writes
            _Float16* stc = sm.st + ((l * 2 + cb) * MROWS + mt * 16) * SDIM;
            #pragma unroll
            for (int r = 0; r < 8; ++r)
                stc[(r + 8 * hf) * SDIM + w4 * 16 + l16] = (_Float16)ps[l][r];
            __syncthreads();

            // ================= output projection: state @ Wo^T + residual =================
            const _Float16* wol = sm.wo + l * DDIM * SDIM;
            v16h sa0  = ld_a(stc, SDIM, 0, lane);
            v16h sa1  = ld_a(stc, SDIM, 1, lane);
            v16h bo00 = ld_b(wol, SDIM, w4 * 2 + 0, 0, lane);
            v16h bo01 = ld_b(wol, SDIM, w4 * 2 + 0, 1, lane);
            v16h bo10 = ld_b(wol, SDIM, w4 * 2 + 1, 0, lane);
            v16h bo11 = ld_b(wol, SDIM, w4 * 2 + 1, 1, lane);
            v8f acco0 = zero8, acco1 = zero8;
            acco0 = WMMA(sa0, bo00, acco0);
            acco1 = WMMA(sa0, bo10, acco1);
            acco0 = WMMA(sa1, bo01, acco0);
            acco1 = WMMA(sa1, bo11, acco1);
            #pragma unroll
            for (int r = 0; r < 8; ++r) {
                const int grow = mt * 16 + r + 8 * hf;
                const int n0 = (w4 * 2 + 0) * 16 + l16;
                const int n1 = (w4 * 2 + 1) * 16 + l16;
                sm.xo[grow * DDIM + n0] = sigf(acco0[r] + wob[l][0]) + sm.xn32[grow * DDIM + n0];
                sm.xo[grow * DDIM + n1] = sigf(acco1[r] + wob[l][1]) + sm.xn32[grow * DDIM + n1];
            }
            __syncthreads();
        } // layers

        // ---- store x_t output (same thread<->element mapping as the load; no race) ----
        float* op = out + xbase + (size_t)t * DDIM;
        const float* xr2 = &sm.xo[row * DDIM + sub * 16];
        #pragma unroll
        for (int i = 0; i < 4; ++i)
            ((float4*)op)[i] = ((const float4*)xr2)[i];
    } // t
}

extern "C" void kernel_launch(void* const* d_in, const int* in_sizes, int n_in,
                              void* d_out, int out_size, void* d_ws, size_t ws_size,
                              hipStream_t stream) {
    (void)in_sizes; (void)n_in; (void)out_size; (void)d_ws; (void)ws_size;
    const float* x_seq = (const float*)d_in[0];
    const float* Wg_w  = (const float*)d_in[1];
    const float* Wg_b  = (const float*)d_in[2];
    const float* Ug_w  = (const float*)d_in[3];
    const float* Ug_b  = (const float*)d_in[4];
    const float* bg    = (const float*)d_in[5];
    const float* Wq_w  = (const float*)d_in[6];
    const float* Wq_b  = (const float*)d_in[7];
    const float* Wv_w  = (const float*)d_in[8];
    const float* Wv_b  = (const float*)d_in[9];
    const float* Avec  = (const float*)d_in[10];
    const float* B_w   = (const float*)d_in[11];
    const float* B_b   = (const float*)d_in[12];
    const float* Wo_w  = (const float*)d_in[13];
    const float* Wo_b  = (const float*)d_in[14];
    const float* ln_g  = (const float*)d_in[15];
    const float* ln_b  = (const float*)d_in[16];
    const float* bn_g  = (const float*)d_in[17];
    const float* bn_b  = (const float*)d_in[18];
    const float* bn_rm = (const float*)d_in[19];
    const float* bn_rv = (const float*)d_in[20];
    float* out = (float*)d_out;

    rmsu_scan_kernel<<<dim3(BATCH / MROWS), dim3(NTHR), sizeof(Smem), stream>>>(
        x_seq, Wg_w, Wg_b, Ug_w, Ug_b, bg, Wq_w, Wq_b, Wv_w, Wv_b,
        Avec, B_w, B_b, Wo_w, Wo_b, ln_g, ln_b, bn_g, bn_b, bn_rm, bn_rv, out);
}
